// GeoNetEuclidean_5007931867346
// MI455X (gfx1250) — matrix operand order repaired and checked by reading, not compiled, not run on hardware
//
#include <hip/hip_runtime.h>
#include <hip/hip_bf16.h>
#include <math.h>

typedef _Float16 f16;
typedef __attribute__((ext_vector_type(16))) _Float16 v16h;
typedef __attribute__((ext_vector_type(8)))  _Float16 v8h;
typedef __attribute__((ext_vector_type(4)))  _Float16 v4h;
typedef __attribute__((ext_vector_type(8)))  float    v8f;

#define NNODES 4096
#define NEDGES 131072
#define HIDDIM 256
#define NHEAD  4
#define NLAYER 3
#define INDIM  128
#define OUTDIM 64

#define FLAG_RELU   1
#define FLAG_F16OUT 2

__device__ __forceinline__ v8f wmma16(v16h a, v16h b, v8f c) {
  return __builtin_amdgcn_wmma_f32_16x16x32_f16(false, a, false, b, (short)0, c,
                                                false, false);
}

// ---------------------------------------------------------------------------
// fp32 -> f16 bulk conversion (float4 in, 4xf16 out per thread).
// ---------------------------------------------------------------------------
__global__ void cvt_f32_f16_kernel(const float* __restrict__ in,
                                   f16* __restrict__ out, int n4) {
  int i = blockIdx.x * 256 + threadIdx.x;
  if (i < n4) {
    float4 v = ((const float4*)in)[i];
    v4h h;
    h[0] = (f16)v.x; h[1] = (f16)v.y; h[2] = (f16)v.z; h[3] = (f16)v.w;
    ((v4h*)out)[i] = h;
  }
}

// ---------------------------------------------------------------------------
// Tiled GEMM: C[M,N] = act(A[M,K] @ W[N,K]^T + bias). A, W are f16.
// 256 threads (8 waves), BM=128, BN=64, BK=32, double-buffered LDS,
// register-staged b128 global loads. M%128==0, N%64==0, K%32==0.
// ---------------------------------------------------------------------------
__global__ __launch_bounds__(256)
void gemm_nt_kernel(const f16* __restrict__ A, const f16* __restrict__ W,
                    const float* __restrict__ bias, void* __restrict__ Cout,
                    int N, int K, int flags) {
  __shared__ __align__(16) f16 As[2][128][40];   // 80B row stride (16B aligned)
  __shared__ __align__(16) f16 Bs[2][64][40];
  const int t    = threadIdx.x;
  const int lane = t & 31;
  const int wave = t >> 5;
  const int wm   = wave & 3;
  const int wn   = wave >> 2;
  const int l15  = lane & 15;
  const int lh   = lane >> 4;
  const int rowBase = blockIdx.y * 128;
  const int colBase = blockIdx.x * 64;

  // Chunk mapping: A tile = 128x32 halfs = 512 v8h chunks (2/thread),
  // W tile = 64x32 halfs = 256 chunks (1/thread). row = c>>2, col = (c&3)*8.
  const int ar0 = t >> 2, ac = (t & 3) * 8;   // chunk t   -> rows 0..63
  const int ar1 = ar0 + 64;                    // chunk t+256 -> rows 64..127
  const int br  = t >> 2, bc = (t & 3) * 8;

  v8f acc[2][2];
#pragma unroll
  for (int i = 0; i < 2; ++i)
#pragma unroll
    for (int j = 0; j < 2; ++j)
#pragma unroll
      for (int r = 0; r < 8; ++r) acc[i][j][r] = 0.0f;

  // Preload first tile into registers (batched b128 loads).
  v8h aR0 = *(const v8h*)&A[(size_t)(rowBase + ar0) * K + ac];
  v8h aR1 = *(const v8h*)&A[(size_t)(rowBase + ar1) * K + ac];
  v8h bR  = *(const v8h*)&W[(size_t)(colBase + br) * K + bc];

  int p = 0;
  for (int k0 = 0; k0 < K; k0 += 32) {
    // Commit staged registers to LDS buffer p.
    *(v8h*)&As[p][ar0][ac] = aR0;
    *(v8h*)&As[p][ar1][ac] = aR1;
    *(v8h*)&Bs[p][br][bc]  = bR;
    __syncthreads();

    // Issue next tile's global loads; they overlap the WMMAs below.
    if (k0 + 32 < K) {
      aR0 = *(const v8h*)&A[(size_t)(rowBase + ar0) * K + k0 + 32 + ac];
      aR1 = *(const v8h*)&A[(size_t)(rowBase + ar1) * K + k0 + 32 + ac];
      bR  = *(const v8h*)&W[(size_t)(colBase + br) * K + k0 + 32 + bc];
    }

    // A fragment: m = lane&15, k = (lane>>4)*16 + j
    v16h af[2];
#pragma unroll
    for (int tm = 0; tm < 2; ++tm)
#pragma unroll
      for (int j = 0; j < 16; ++j)
        af[tm][j] = As[p][wm * 32 + tm * 16 + l15][(lh << 4) + j];

    // B fragment: n = lane&15, k = (lane>>4)*16 + j
    v16h bf[2];
#pragma unroll
    for (int tn = 0; tn < 2; ++tn)
#pragma unroll
      for (int j = 0; j < 16; ++j)
        bf[tn][j] = Bs[p][wn * 32 + tn * 16 + l15][(lh << 4) + j];

#pragma unroll
    for (int tm = 0; tm < 2; ++tm)
#pragma unroll
      for (int tn = 0; tn < 2; ++tn)
        acc[tm][tn] = wmma16(af[tm], bf[tn], acc[tm][tn]);

    p ^= 1;  // single barrier per iter is safe with 2 LDS buffers
  }

  // Epilogue. C layout: m = (lane>>4)*8 + r, n = lane&15.
#pragma unroll
  for (int tm = 0; tm < 2; ++tm)
#pragma unroll
    for (int tn = 0; tn < 2; ++tn)
#pragma unroll
      for (int r = 0; r < 8; ++r) {
        int row = rowBase + wm * 32 + tm * 16 + lh * 8 + r;
        int col = colBase + wn * 32 + tn * 16 + l15;
        float v = acc[tm][tn][r];
        if (bias) v += bias[col];
        if (flags & FLAG_RELU) v = fmaxf(v, 0.0f);
        if (flags & FLAG_F16OUT)
          ((f16*)Cout)[(size_t)row * N + col] = (f16)v;
        else
          ((float*)Cout)[(size_t)row * N + col] = v;
      }
}

// ---------------------------------------------------------------------------
// Flash attention over f16 qkv[N][768]: block = (head, 128-row Q block),
// 8 waves x 16 query rows. 64-row K/V tiles, online softmax, f16 WMMA,
// fp32 stats/accumulators, f16 output. Next K/V tile register-prefetched.
// ---------------------------------------------------------------------------
__global__ __launch_bounds__(256)
void flash_attn_kernel(const f16* __restrict__ qkv, f16* __restrict__ o) {
  __shared__ __align__(16) f16 Qs[128][72];     // 144B stride, 16B aligned
  __shared__ __align__(16) f16 Ks[64][72];
  __shared__ __align__(16) f16 Vs[64][72];
  __shared__ __align__(16) f16 Ps[8][16][64];

  const int t    = threadIdx.x;
  const int lane = t & 31;
  const int wave = t >> 5;
  const int l15  = lane & 15;
  const int lh   = lane >> 4;
  const int head = blockIdx.y;
  const int q0   = blockIdx.x * 128;
  const float scale = 0.125f;                   // 1/sqrt(64)

  // Q tile: 128x64 halfs = 1024 v8h chunks, 4 per thread.
#pragma unroll
  for (int i = 0; i < 4; ++i) {
    int c = t + i * 256;
    int r = c >> 3, col = (c & 7) * 8;
    *(v8h*)&Qs[r][col] =
        *(const v8h*)&qkv[(size_t)(q0 + r) * 768 + head * 64 + col];
  }

  // K/V chunk mapping: 64x64 halfs = 512 chunks, 2 per thread each.
  const int r0 = t >> 3, c0 = (t & 7) * 8;      // chunk t   -> rows 0..31
  const int r1 = r0 + 32;                        // chunk t+256 -> rows 32..63
  v8h kR0 = *(const v8h*)&qkv[(size_t)r0 * 768 + 256 + head * 64 + c0];
  v8h kR1 = *(const v8h*)&qkv[(size_t)r1 * 768 + 256 + head * 64 + c0];
  v8h vR0 = *(const v8h*)&qkv[(size_t)r0 * 768 + 512 + head * 64 + c0];
  v8h vR1 = *(const v8h*)&qkv[(size_t)r1 * 768 + 512 + head * 64 + c0];

  v8f oacc[4];
  float rmax[8], rsum[8];
#pragma unroll
  for (int nt = 0; nt < 4; ++nt)
#pragma unroll
    for (int r = 0; r < 8; ++r) oacc[nt][r] = 0.0f;
#pragma unroll
  for (int r = 0; r < 8; ++r) { rmax[r] = -1e30f; rsum[r] = 0.0f; }

  __syncthreads();  // Q visible

  for (int kv0 = 0; kv0 < NNODES; kv0 += 64) {
    // Commit staged K/V registers to LDS.
    *(v8h*)&Ks[r0][c0] = kR0;
    *(v8h*)&Ks[r1][c0] = kR1;
    *(v8h*)&Vs[r0][c0] = vR0;
    *(v8h*)&Vs[r1][c0] = vR1;
    __syncthreads();

    // Prefetch next K/V tile into registers (overlaps compute below).
    if (kv0 + 64 < NNODES) {
      size_t base0 = (size_t)(kv0 + 64 + r0) * 768 + head * 64 + c0;
      size_t base1 = (size_t)(kv0 + 64 + r1) * 768 + head * 64 + c0;
      kR0 = *(const v8h*)&qkv[base0 + 256];
      kR1 = *(const v8h*)&qkv[base1 + 256];
      vR0 = *(const v8h*)&qkv[base0 + 512];
      vR1 = *(const v8h*)&qkv[base1 + 512];
    }

    // S (16x64 per wave) = Q @ K^T, contract dh=64 in 2 steps of 32.
    v8f s[4];
#pragma unroll
    for (int nt = 0; nt < 4; ++nt)
#pragma unroll
      for (int r = 0; r < 8; ++r) s[nt][r] = 0.0f;
#pragma unroll
    for (int ks = 0; ks < 2; ++ks) {
      v16h aq;
#pragma unroll
      for (int j = 0; j < 16; ++j)
        aq[j] = Qs[wave * 16 + l15][ks * 32 + (lh << 4) + j];
#pragma unroll
      for (int nt = 0; nt < 4; ++nt) {
        v16h bk;  // B[k_dh][n_kv] = K[n_kv][k_dh]
#pragma unroll
        for (int j = 0; j < 16; ++j)
          bk[j] = Ks[nt * 16 + l15][ks * 32 + (lh << 4) + j];
        s[nt] = wmma16(aq, bk, s[nt]);
      }
    }

    // Online softmax; row m=(lane>>4)*8+r lives in 16 lanes of one half-wave.
#pragma unroll
    for (int r = 0; r < 8; ++r) {
      float m = -1e30f;
#pragma unroll
      for (int nt = 0; nt < 4; ++nt) m = fmaxf(m, s[nt][r]);
      m *= scale;
      m = fmaxf(m, __shfl_xor(m, 1, 32));
      m = fmaxf(m, __shfl_xor(m, 2, 32));
      m = fmaxf(m, __shfl_xor(m, 4, 32));
      m = fmaxf(m, __shfl_xor(m, 8, 32));
      float mnew  = fmaxf(rmax[r], m);
      float alpha = __expf(rmax[r] - mnew);
      float lsum  = 0.0f;
#pragma unroll
      for (int nt = 0; nt < 4; ++nt) {
        float pv = __expf(s[nt][r] * scale - mnew);
        lsum += pv;
        Ps[wave][lh * 8 + r][nt * 16 + l15] = (f16)pv;
      }
      lsum += __shfl_xor(lsum, 1, 32);
      lsum += __shfl_xor(lsum, 2, 32);
      lsum += __shfl_xor(lsum, 4, 32);
      lsum += __shfl_xor(lsum, 8, 32);
      rsum[r] = rsum[r] * alpha + lsum;
      rmax[r] = mnew;
#pragma unroll
      for (int nt = 0; nt < 4; ++nt) oacc[nt][r] *= alpha;
    }

    // O += P(16x64) @ V(64x64), contract kv in 2 steps of 32.
#pragma unroll
    for (int ks = 0; ks < 2; ++ks) {
      v16h ap;
#pragma unroll
      for (int j = 0; j < 16; ++j)
        ap[j] = Ps[wave][l15][ks * 32 + (lh << 4) + j];
#pragma unroll
      for (int nt = 0; nt < 4; ++nt) {
        v16h bv;  // B[k_kv][n_dh] = V[k_kv][n_dh]
#pragma unroll
        for (int j = 0; j < 16; ++j)
          bv[j] = Vs[ks * 32 + (lh << 4) + j][nt * 16 + l15];
        oacc[nt] = wmma16(ap, bv, oacc[nt]);
      }
    }
    __syncthreads();  // all waves done with Ks/Vs before next commit
  }

  // Normalize, write f16 o[row][head*64+col].
#pragma unroll
  for (int nt = 0; nt < 4; ++nt)
#pragma unroll
    for (int r = 0; r < 8; ++r) {
      int row = q0 + wave * 16 + lh * 8 + r;
      int col = nt * 16 + l15;
      o[(size_t)row * HIDDIM + head * 64 + col] = (f16)(oacc[nt][r] / rsum[r]);
    }
}

// ---------------------------------------------------------------------------
// Glue kernels
// ---------------------------------------------------------------------------
__global__ void deg_kernel(const int* __restrict__ dst, float* __restrict__ deg) {
  int e = blockIdx.x * 256 + threadIdx.x;
  if (e < NEDGES) atomicAdd(&deg[dst[e]], 1.0f);
}

__global__ void dis_kernel(const float* __restrict__ deg, float* __restrict__ dis) {
  int i = blockIdx.x * 256 + threadIdx.x;
  if (i < NNODES) dis[i] = rsqrtf(deg[i] + 1.0f);
}

__global__ void rowscale_kernel(const float* __restrict__ h,
                                const float* __restrict__ dis,
                                float* __restrict__ out) {
  int idx = blockIdx.x * 256 + threadIdx.x;
  out[idx] = h[idx] * dis[idx >> 8];
}

__global__ void addscale_kernel(const float* __restrict__ agg,
                                const float* __restrict__ h,
                                const float* __restrict__ dis,
                                float* __restrict__ out) {
  int idx = blockIdx.x * 256 + threadIdx.x;
  out[idx] = (agg[idx] + h[idx]) * dis[idx >> 8];
}

__global__ void scatter_kernel(const int* __restrict__ src,
                               const int* __restrict__ dst,
                               const float* __restrict__ hs,
                               float* __restrict__ agg) {
  int tid = blockIdx.x * 256 + threadIdx.x;   // E*64 threads, 4 floats each
  int e = tid >> 6;
  int f = (tid & 63) << 2;
  int s = src[e], d = dst[e];
  const float4 v = *(const float4*)&hs[(size_t)s * HIDDIM + f];
  atomicAdd(&agg[(size_t)d * HIDDIM + f + 0], v.x);
  atomicAdd(&agg[(size_t)d * HIDDIM + f + 1], v.y);
  atomicAdd(&agg[(size_t)d * HIDDIM + f + 2], v.z);
  atomicAdd(&agg[(size_t)d * HIDDIM + f + 3], v.w);
}

__global__ __launch_bounds__(256)
void residual_ln_kernel(float* __restrict__ h, const float* __restrict__ hc,
                        const float* __restrict__ ha, const float* __restrict__ g,
                        const float* __restrict__ b) {
  __shared__ float red[256];
  int row = blockIdx.x, t = threadIdx.x;
  size_t i = (size_t)row * HIDDIM + t;
  float z = h[i] + hc[i] + ha[i];
  red[t] = z;
  __syncthreads();
#pragma unroll
  for (int s = 128; s > 0; s >>= 1) {
    if (t < s) red[t] += red[t + s];
    __syncthreads();
  }
  float mu = red[0] * (1.0f / HIDDIM);
  __syncthreads();
  float d = z - mu;
  red[t] = d * d;
  __syncthreads();
#pragma unroll
  for (int s = 128; s > 0; s >>= 1) {
    if (t < s) red[t] += red[t + s];
    __syncthreads();
  }
  float var = red[0] * (1.0f / HIDDIM);
  h[i] = d * rsqrtf(var + 1e-5f) * g[t] + b[t];
}

// ---------------------------------------------------------------------------
extern "C" void kernel_launch(void* const* d_in, const int* in_sizes, int n_in,
                              void* d_out, int out_size, void* d_ws, size_t ws_size,
                              hipStream_t stream) {
  (void)in_sizes; (void)n_in; (void)out_size; (void)ws_size;
  const float* x          = (const float*)d_in[0];
  const int*   edge_index = (const int*)d_in[1];
  const float* enc_w      = (const float*)d_in[2];
  const float* enc_b      = (const float*)d_in[3];
  const float* gcn_w      = (const float*)d_in[4];
  const float* in_proj_w  = (const float*)d_in[5];
  const float* in_proj_b  = (const float*)d_in[6];
  const float* out_proj_w = (const float*)d_in[7];
  const float* out_proj_b = (const float*)d_in[8];
  const float* ln_g       = (const float*)d_in[9];
  const float* ln_b       = (const float*)d_in[10];
  const float* head_w     = (const float*)d_in[11];
  const float* head_b     = (const float*)d_in[12];
  const int* srcp = edge_index;
  const int* dstp = edge_index + NEDGES;

  // Workspace carving (256B-aligned chunks).
  char* cur = (char*)d_ws;
  auto alloc = [&](size_t bytes) {
    char* r = cur;
    cur += (bytes + 255) & ~(size_t)255;
    return r;
  };
  const size_t NH = (size_t)NNODES * HIDDIM;               // 1M elems
  float* deg   = (float*)alloc(NNODES * 4);
  float* dis   = (float*)alloc(NNODES * 4);
  float* h     = (float*)alloc(NH * 4);
  float* gin   = (float*)alloc(NH * 4);                    // hs, then GCN input
  float* agg   = (float*)alloc(NH * 4);
  float* hc    = (float*)alloc(NH * 4);
  float* ha    = (float*)alloc(NH * 4);
  f16* x16     = (f16*)alloc((size_t)NNODES * INDIM * 2);
  f16* h16     = (f16*)alloc(NH * 2);
  f16* gin16   = (f16*)alloc(NH * 2);
  f16* qkv16   = (f16*)alloc((size_t)NNODES * 768 * 2);
  f16* ov16    = (f16*)alloc(NH * 2);
  f16* enc_w16 = (f16*)alloc((size_t)HIDDIM * INDIM * 2);
  f16* gcn_w16 = (f16*)alloc((size_t)NLAYER * HIDDIM * HIDDIM * 2);
  f16* ipw16   = (f16*)alloc((size_t)NLAYER * 768 * HIDDIM * 2);
  f16* opw16   = (f16*)alloc((size_t)NLAYER * HIDDIM * HIDDIM * 2);
  f16* hdw16   = (f16*)alloc((size_t)OUTDIM * HIDDIM * 2);

  dim3 blk(256);
  auto cvt = [&](const float* in, f16* out, size_t n) {
    int n4 = (int)(n / 4);
    cvt_f32_f16_kernel<<<(n4 + 255) / 256, blk, 0, stream>>>(in, out, n4);
  };

  // One-time fp32 -> f16 conversions (weights + encoder input).
  cvt(x, x16, (size_t)NNODES * INDIM);
  cvt(enc_w, enc_w16, (size_t)HIDDIM * INDIM);
  cvt(gcn_w, gcn_w16, (size_t)NLAYER * HIDDIM * HIDDIM);
  cvt(in_proj_w, ipw16, (size_t)NLAYER * 768 * HIDDIM);
  cvt(out_proj_w, opw16, (size_t)NLAYER * HIDDIM * HIDDIM);
  cvt(head_w, hdw16, (size_t)OUTDIM * HIDDIM);

  // Degree and normalization.
  hipMemsetAsync(deg, 0, NNODES * sizeof(float), stream);
  deg_kernel<<<NEDGES / 256, blk, 0, stream>>>(dstp, deg);
  dis_kernel<<<NNODES / 256, blk, 0, stream>>>(deg, dis);

  // Encoder: h = relu(x @ enc_w^T + enc_b)
  gemm_nt_kernel<<<dim3(HIDDIM / 64, NNODES / 128), blk, 0, stream>>>(
      x16, enc_w16, enc_b, h, HIDDIM, INDIM, FLAG_RELU);

  for (int l = 0; l < NLAYER; ++l) {
    // GCN branch.
    rowscale_kernel<<<(int)(NH / 256), blk, 0, stream>>>(h, dis, gin);
    hipMemsetAsync(agg, 0, NH * sizeof(float), stream);
    scatter_kernel<<<(NEDGES * 64) / 256, blk, 0, stream>>>(srcp, dstp, gin, agg);
    addscale_kernel<<<(int)(NH / 256), blk, 0, stream>>>(agg, h, dis, gin);
    cvt(gin, gin16, NH);
    gemm_nt_kernel<<<dim3(HIDDIM / 64, NNODES / 128), blk, 0, stream>>>(
        gin16, gcn_w16 + (size_t)l * HIDDIM * HIDDIM, nullptr, hc,
        HIDDIM, HIDDIM, FLAG_RELU);

    // Attention branch (f16 end-to-end: qkv -> flash -> out_proj input).
    cvt(h, h16, NH);
    gemm_nt_kernel<<<dim3(768 / 64, NNODES / 128), blk, 0, stream>>>(
        h16, ipw16 + (size_t)l * 768 * HIDDIM, in_proj_b + (size_t)l * 768,
        qkv16, 768, HIDDIM, FLAG_F16OUT);
    flash_attn_kernel<<<dim3(NNODES / 128, NHEAD), blk, 0, stream>>>(qkv16, ov16);
    gemm_nt_kernel<<<dim3(HIDDIM / 64, NNODES / 128), blk, 0, stream>>>(
        ov16, opw16 + (size_t)l * HIDDIM * HIDDIM, out_proj_b + (size_t)l * HIDDIM,
        ha, HIDDIM, HIDDIM, 0);

    // Residual + LayerNorm (in place on h).
    residual_ln_kernel<<<NNODES, blk, 0, stream>>>(
        h, hc, ha, ln_g + (size_t)l * HIDDIM, ln_b + (size_t)l * HIDDIM);
  }

  // Head: out = h @ head_w^T + head_b
  cvt(h, h16, NH);
  gemm_nt_kernel<<<dim3(OUTDIM / 64, NNODES / 128), blk, 0, stream>>>(
      h16, hdw16, head_b, (float*)d_out, OUTDIM, HIDDIM, 0);
}
